// EmbeddingAttentionLSTMRegressor_89163521065567
// MI455X (gfx1250) — matrix operand order repaired and checked by reading, not compiled, not run on hardware
//
#include <hip/hip_runtime.h>
#include <hip/hip_bf16.h>

// ---------------------------------------------------------------------------
// EmbeddingAttentionLSTMRegressor for MI455X (gfx1250, wave32, WMMA)
//   B=64, T=256, Din=128, U=1024, 4U=4096, SEQ=256
// GEMMs in bf16 via v_wmma_f32_16x16x32_bf16 with 32x32 register blocking
// (4 accumulators/wave -> 2 b128 loads per WMMA); softmax / tanh / LSTM
// gates in f32 VALU (native v_tanh/v_exp transcendentals). Whole working
// set (~120 MB) is L2-resident on the 192 MB L2.
// ---------------------------------------------------------------------------

typedef __attribute__((ext_vector_type(8)))  __bf16 v8bf;
typedef __attribute__((ext_vector_type(16))) __bf16 v16bf;
typedef __attribute__((ext_vector_type(8)))  float  v8f;

#define BB   64
#define TT   256
#define DIN  128
#define UU   1024
#define ZN   4096
#define SEQ  256

// ---- WMMA fragment loaders (wave32 layouts per CDNA5 ISA 7.12.2) ----------
// A 16x32 bf16, row-major src, stride = ld elements.
//   lane<16 : row=lane,    halves 0..7 = K 0..7,  halves 8..15 = K 16..23
//   lane>=16: row=lane-16, halves 0..7 = K 8..15, halves 8..15 = K 24..31
__device__ __forceinline__ v16bf load_frag_a(const __bf16* base, int ld) {
    int lane = threadIdx.x & 31;
    const __bf16* p = base + (lane & 15) * ld + ((lane >> 4) << 3);
    v8bf lo = *(const v8bf*)(p);
    v8bf hi = *(const v8bf*)(p + 16);
    v16bf a;
#pragma unroll
    for (int i = 0; i < 8; ++i) { a[i] = lo[i]; a[i + 8] = hi[i]; }
    return a;
}

// B 32x16 bf16 taken from a TRANSPOSED (N x K row-major) weight matrix:
//   lane<16 : col=lane,    halves 0..7 = K 0..7,  halves 8..15 = K 16..23
//   lane>=16: col=lane-16, halves 0..7 = K 8..15, halves 8..15 = K 24..31
__device__ __forceinline__ v16bf load_frag_b(const __bf16* baseT, int ld) {
    int lane = threadIdx.x & 31;
    const __bf16* p = baseT + (lane & 15) * ld + ((lane >> 4) << 3);
    v8bf lo = *(const v8bf*)(p);
    v8bf hi = *(const v8bf*)(p + 16);
    v16bf b;
#pragma unroll
    for (int i = 0; i < 8; ++i) { b[i] = lo[i]; b[i + 8] = hi[i]; }
    return b;
}

// Store one 16x16 f32 C tile (VGPR r -> row r / r+8 per half-wave).
template <bool RELU>
__device__ __forceinline__ void store_tile(const v8f& acc, const float* bias,
                                           float* Cf, __bf16* Cbf,
                                           int m0, int n0, int N) {
    int lane = threadIdx.x & 31;
    int n    = n0 + (lane & 15);
    int mb   = m0 + ((lane >> 4) << 3);
    float bv = bias ? bias[n] : 0.0f;
#pragma unroll
    for (int r = 0; r < 8; ++r) {
        float v = acc[r] + bv;
        if (RELU) v = fmaxf(v, 0.0f);
        size_t idx = (size_t)(mb + r) * N + n;
        if (Cf)  Cf[idx]  = v;
        if (Cbf) Cbf[idx] = (__bf16)v;
    }
}

// ---- bf16 WMMA GEMM: C[M,N] = act(A@B (+ A2@B2) + bias) -------------------
// A row-major [M,K]; Bt row-major TRANSPOSED weights [N,K]. One wave per
// 32x32 C block: 4 accumulators, 2 A-frag + 2 B-frag loads per K step ->
// 2 b128 loads per WMMA. All dims multiples of 32; K multiples of 32.
template <bool RELU, bool DUAL>
__global__ void wmma_gemm_kernel(const __bf16* __restrict__ A,
                                 const __bf16* __restrict__ Bt,
                                 const __bf16* __restrict__ A2,
                                 const __bf16* __restrict__ B2t,
                                 const float*  __restrict__ bias,
                                 float*        __restrict__ Cf,
                                 __bf16*       __restrict__ Cbf,
                                 int M, int N, int K, int K2) {
    int wid    = (blockIdx.x * blockDim.x + threadIdx.x) >> 5;
    int tilesN = N >> 5;
    int total  = (M >> 5) * tilesN;
    if (wid >= total) return;                 // wave-uniform: EXEC stays full
    int m0 = (wid / tilesN) << 5;
    int n0 = (wid % tilesN) << 5;

    v8f acc00 = {}, acc01 = {}, acc10 = {}, acc11 = {};

    {
        const __bf16* ap0 = A  + (size_t)m0 * K;
        const __bf16* ap1 = A  + (size_t)(m0 + 16) * K;
        const __bf16* bp0 = Bt + (size_t)n0 * K;
        const __bf16* bp1 = Bt + (size_t)(n0 + 16) * K;
        for (int k0 = 0; k0 < K; k0 += 32) {
            v16bf a0 = load_frag_a(ap0 + k0, K);
            v16bf a1 = load_frag_a(ap1 + k0, K);
            v16bf b0 = load_frag_b(bp0 + k0, K);
            v16bf b1 = load_frag_b(bp1 + k0, K);
            acc00 = __builtin_amdgcn_wmma_f32_16x16x32_bf16(false, a0, false, b0, (short)0, acc00, false, false);
            acc01 = __builtin_amdgcn_wmma_f32_16x16x32_bf16(false, a0, false, b1, (short)0, acc01, false, false);
            acc10 = __builtin_amdgcn_wmma_f32_16x16x32_bf16(false, a1, false, b0, (short)0, acc10, false, false);
            acc11 = __builtin_amdgcn_wmma_f32_16x16x32_bf16(false, a1, false, b1, (short)0, acc11, false, false);
        }
    }
    if (DUAL) {
        const __bf16* ap0 = A2  + (size_t)m0 * K2;
        const __bf16* ap1 = A2  + (size_t)(m0 + 16) * K2;
        const __bf16* bp0 = B2t + (size_t)n0 * K2;
        const __bf16* bp1 = B2t + (size_t)(n0 + 16) * K2;
        for (int k0 = 0; k0 < K2; k0 += 32) {
            v16bf a0 = load_frag_a(ap0 + k0, K2);
            v16bf a1 = load_frag_a(ap1 + k0, K2);
            v16bf b0 = load_frag_b(bp0 + k0, K2);
            v16bf b1 = load_frag_b(bp1 + k0, K2);
            acc00 = __builtin_amdgcn_wmma_f32_16x16x32_bf16(false, a0, false, b0, (short)0, acc00, false, false);
            acc01 = __builtin_amdgcn_wmma_f32_16x16x32_bf16(false, a0, false, b1, (short)0, acc01, false, false);
            acc10 = __builtin_amdgcn_wmma_f32_16x16x32_bf16(false, a1, false, b0, (short)0, acc10, false, false);
            acc11 = __builtin_amdgcn_wmma_f32_16x16x32_bf16(false, a1, false, b1, (short)0, acc11, false, false);
        }
    }

    store_tile<RELU>(acc00, bias, Cf, Cbf, m0,      n0,      N);
    store_tile<RELU>(acc01, bias, Cf, Cbf, m0,      n0 + 16, N);
    store_tile<RELU>(acc10, bias, Cf, Cbf, m0 + 16, n0,      N);
    store_tile<RELU>(acc11, bias, Cf, Cbf, m0 + 16, n0 + 16, N);
}

// ---- Elementwise conversion / transpose / init ----------------------------
__global__ void cvt_f32_to_bf16(const float* __restrict__ in,
                                __bf16* __restrict__ out, int n) {
    int i = blockIdx.x * blockDim.x + threadIdx.x;
    if (i < n) out[i] = (__bf16)in[i];
}

// W [R,C] f32 row-major -> Wt [C,R] bf16 row-major
__global__ void transpose_to_bf16(const float* __restrict__ W,
                                  __bf16* __restrict__ Wt, int R, int C) {
    int i = blockIdx.x * blockDim.x + threadIdx.x;
    if (i < R * C) {
        int c = i / R, r = i % R;
        Wt[i] = (__bf16)W[(size_t)r * C + c];
    }
}

__global__ void zero_state(float* __restrict__ h, float* __restrict__ c,
                           __bf16* __restrict__ hbf, int n) {
    int i = blockIdx.x * blockDim.x + threadIdx.x;
    if (i < n) { h[i] = 0.0f; c[i] = 0.0f; hbf[i] = (__bf16)0.0f; }
}

// ---- Attention: score=tanh(q+keys)@V, softmax over T, ctx=attw@x ----------
// One block (1024 threads = 32 waves) per batch row. keys/x stay L2-resident.
__global__ void attn_ctx_kernel(const float* __restrict__ q,
                                const float* __restrict__ keys,
                                const __bf16* __restrict__ xbf,
                                const float* __restrict__ V,
                                const float* __restrict__ bV,
                                float* __restrict__ ctx,
                                __bf16* __restrict__ ctxbf) {
    __shared__ float sq[UU];
    __shared__ float sv[UU];
    __shared__ float ssc[TT];
    __shared__ float ssum;

    int b = blockIdx.x, tid = threadIdx.x;
    sq[tid] = q[b * UU + tid];
    sv[tid] = V[tid];
    __syncthreads();

    int wv = tid >> 5, lane = tid & 31;
    const float* kb = keys + (size_t)b * TT * UU;
    for (int t = wv; t < TT; t += 32) {          // 32 waves x 8 rows each
        const float* kr = kb + (size_t)t * UU;
        float p = 0.0f;
        for (int u = lane; u < UU; u += 32)
            p += tanhf(sq[u] + kr[u]) * sv[u];
#pragma unroll
        for (int off = 16; off; off >>= 1) p += __shfl_xor(p, off);
        if (lane == 0) ssc[t] = p + bV[0];
    }
    __syncthreads();

    if (tid < 32) {                               // single-wave softmax stats
        float mx = -3.402823466e38f;
        for (int t = lane; t < TT; t += 32) mx = fmaxf(mx, ssc[t]);
#pragma unroll
        for (int off = 16; off; off >>= 1) mx = fmaxf(mx, __shfl_xor(mx, off));
        float s = 0.0f;
        for (int t = lane; t < TT; t += 32) {
            float e = __expf(ssc[t] - mx);
            ssc[t] = e;                            // unnormalized weight
            s += e;
        }
#pragma unroll
        for (int off = 16; off; off >>= 1) s += __shfl_xor(s, off);
        if (lane == 0) ssum = s;
    }
    __syncthreads();

    float inv = 1.0f / ssum;
    const __bf16* xb = xbf + (size_t)b * TT * UU;
    float c = 0.0f;
    for (int t = 0; t < TT; ++t)                  // lane-contiguous in u
        c += ssc[t] * (float)xb[(size_t)t * UU + tid];
    c *= inv;
    ctx[b * UU + tid]   = c;
    ctxbf[b * UU + tid] = (__bf16)c;
}

// ---- LSTM gates (Keras order i,f,g,o) -------------------------------------
__global__ void lstm_gates_kernel(const float* __restrict__ z,
                                  float* __restrict__ c,
                                  float* __restrict__ h,
                                  __bf16* __restrict__ hbf) {
    int i = blockIdx.x * blockDim.x + threadIdx.x;   // 64*1024
    int b = i >> 10, u = i & (UU - 1);
    const float* zr = z + (size_t)b * ZN;
    float ig = 1.0f / (1.0f + __expf(-zr[u]));
    float fg = 1.0f / (1.0f + __expf(-zr[UU + u]));
    float gg = tanhf(zr[2 * UU + u]);
    float og = 1.0f / (1.0f + __expf(-zr[3 * UU + u]));
    float cn = fg * c[i] + ig * gg;
    float hn = og * tanhf(cn);
    c[i] = cn;
    h[i] = hn;
    hbf[i] = (__bf16)hn;
}

// ---- out[t, b] = h_new[b] @ Wo + bo ---------------------------------------
__global__ void out_proj_kernel(const float* __restrict__ h,
                                const float* __restrict__ Wo,
                                const float* __restrict__ bo,
                                float* __restrict__ out, int t) {
    __shared__ float sp[8];
    int b = blockIdx.x, tid = threadIdx.x;           // 256 threads
    float p = 0.0f;
    for (int u = tid; u < UU; u += 256) p += h[b * UU + u] * Wo[u];
#pragma unroll
    for (int off = 16; off; off >>= 1) p += __shfl_xor(p, off);
    if ((tid & 31) == 0) sp[tid >> 5] = p;
    __syncthreads();
    if (tid == 0) {
        float s = 0.0f;
#pragma unroll
        for (int w = 0; w < 8; ++w) s += sp[w];
        out[t * BB + b] = s + bo[0];
    }
}

// ---------------------------------------------------------------------------
static inline char* ws_take(char** cur, size_t bytes) {
    char* p = *cur;
    *cur += (bytes + 255) & ~(size_t)255;
    return p;
}

static inline void launch_gemm(bool relu, bool dual,
                               const __bf16* A, const __bf16* Bt,
                               const __bf16* A2, const __bf16* B2t,
                               const float* bias, float* Cf, __bf16* Cbf,
                               int M, int N, int K, int K2, hipStream_t s) {
    int waves   = (M >> 5) * (N >> 5);   // one wave per 32x32 block
    int threads = waves * 32;
    int blocks  = (threads + 255) / 256;
    if (dual)
        wmma_gemm_kernel<false, true><<<blocks, 256, 0, s>>>(A, Bt, A2, B2t, bias, Cf, Cbf, M, N, K, K2);
    else if (relu)
        wmma_gemm_kernel<true, false><<<blocks, 256, 0, s>>>(A, Bt, A2, B2t, bias, Cf, Cbf, M, N, K, K2);
    else
        wmma_gemm_kernel<false, false><<<blocks, 256, 0, s>>>(A, Bt, A2, B2t, bias, Cf, Cbf, M, N, K, K2);
}

extern "C" void kernel_launch(void* const* d_in, const int* in_sizes, int n_in,
                              void* d_out, int out_size, void* d_ws, size_t ws_size,
                              hipStream_t stream) {
    (void)in_sizes; (void)n_in; (void)out_size; (void)ws_size;
    const float* inputs = (const float*)d_in[0];
    const float* We1 = (const float*)d_in[1];  const float* be1 = (const float*)d_in[2];
    const float* We2 = (const float*)d_in[3];  const float* be2 = (const float*)d_in[4];
    const float* W1  = (const float*)d_in[5];  const float* b1  = (const float*)d_in[6];
    const float* W2  = (const float*)d_in[7];  const float* b2  = (const float*)d_in[8];
    const float* V   = (const float*)d_in[9];  const float* bV  = (const float*)d_in[10];
    const float* Wk  = (const float*)d_in[11]; const float* Wr  = (const float*)d_in[12];
    const float* bl  = (const float*)d_in[13];
    const float* Wo  = (const float*)d_in[14]; const float* bo  = (const float*)d_in[15];
    float* out = (float*)d_out;

    // ---- workspace carve-out (~123 MB) ----
    char* cur = (char*)d_ws;
    __bf16* we1t  = (__bf16*)ws_take(&cur, (size_t)UU * DIN * 2);   // [U,Din]
    __bf16* we2t  = (__bf16*)ws_take(&cur, (size_t)UU * UU * 2);
    __bf16* w1t   = (__bf16*)ws_take(&cur, (size_t)UU * UU * 2);
    __bf16* w2t   = (__bf16*)ws_take(&cur, (size_t)UU * UU * 2);
    __bf16* wkt   = (__bf16*)ws_take(&cur, (size_t)ZN * UU * 2);    // [4U,U]
    __bf16* wrt   = (__bf16*)ws_take(&cur, (size_t)ZN * UU * 2);
    __bf16* in_bf = (__bf16*)ws_take(&cur, (size_t)BB * TT * DIN * 2);
    __bf16* x_bf  = (__bf16*)ws_take(&cur, (size_t)BB * TT * UU * 2);
    float*  keys  = (float*) ws_take(&cur, (size_t)BB * TT * UU * 4);
    __bf16* h1bf  = (__bf16*)keys;        // alias: h1 dead before keys written
    float*  q     = (float*) ws_take(&cur, (size_t)BB * UU * 4);
    float*  h     = (float*) ws_take(&cur, (size_t)BB * UU * 4);
    float*  c     = (float*) ws_take(&cur, (size_t)BB * UU * 4);
    __bf16* hbf   = (__bf16*)ws_take(&cur, (size_t)BB * UU * 2);
    float*  ctx   = (float*) ws_take(&cur, (size_t)BB * UU * 4);
    __bf16* ctxbf = (__bf16*)ws_take(&cur, (size_t)BB * UU * 2);
    float*  z     = (float*) ws_take(&cur, (size_t)BB * ZN * 4);

    // ---- one-time conversions (per call; deterministic) ----
    {
        int n = BB * TT * DIN;
        cvt_f32_to_bf16<<<(n + 255) / 256, 256, 0, stream>>>(inputs, in_bf, n);
    }
    transpose_to_bf16<<<(DIN * UU + 255) / 256, 256, 0, stream>>>(We1, we1t, DIN, UU);
    transpose_to_bf16<<<(UU * UU + 255) / 256, 256, 0, stream>>>(We2, we2t, UU, UU);
    transpose_to_bf16<<<(UU * UU + 255) / 256, 256, 0, stream>>>(W1,  w1t,  UU, UU);
    transpose_to_bf16<<<(UU * UU + 255) / 256, 256, 0, stream>>>(W2,  w2t,  UU, UU);
    transpose_to_bf16<<<(UU * ZN + 255) / 256, 256, 0, stream>>>(Wk,  wkt,  UU, ZN);
    transpose_to_bf16<<<(UU * ZN + 255) / 256, 256, 0, stream>>>(Wr,  wrt,  UU, ZN);
    zero_state<<<(BB * UU + 255) / 256, 256, 0, stream>>>(h, c, hbf, BB * UU);

    // ---- encoder (WMMA, parallel over B*T rows) ----
    const int M = BB * TT;
    // h1 = relu(in @ We1 + be1)            -> bf16 only
    launch_gemm(true,  false, in_bf, we1t, nullptr, nullptr, be1, nullptr, h1bf, M, UU, DIN, 0, stream);
    // x  = relu(h1 @ We2 + be2)            -> bf16 only (ctx accumulates from bf16 x)
    launch_gemm(true,  false, h1bf,  we2t, nullptr, nullptr, be2, nullptr, x_bf, M, UU, UU, 0, stream);
    // keys = x @ W2 + b2                   -> f32 (feeds tanh/softmax)
    launch_gemm(false, false, x_bf,  w2t,  nullptr, nullptr, b2,  keys, nullptr, M, UU, UU, 0, stream);

    // ---- sequential decode: 256 steps, all on stream ----
    for (int t = 0; t < SEQ; ++t) {
        // q = h @ W1 + b1
        launch_gemm(false, false, hbf, w1t, nullptr, nullptr, b1, q, nullptr, BB, UU, UU, 0, stream);
        // attention + context (L2-resident keys/x)
        attn_ctx_kernel<<<BB, 1024, 0, stream>>>(q, keys, x_bf, V, bV, ctx, ctxbf);
        // z = ctx @ Wk + h @ Wr + bl  (fused dual WMMA)
        launch_gemm(false, true, ctxbf, wkt, hbf, wrt, bl, z, nullptr, BB, ZN, UU, UU, stream);
        // gates -> c, h (f32 + bf16 shadow)
        lstm_gates_kernel<<<(BB * UU + 255) / 256, 256, 0, stream>>>(z, c, h, hbf);
        // out[t, :] = h @ Wo + bo
        out_proj_kernel<<<BB, 256, 0, stream>>>(h, Wo, bo, out, t);
    }
}
